// Style2ResidualBlock1DBeta_35038343201603
// MI455X (gfx1250) — matrix operand order repaired and checked by reading, not compiled, not run on hardware
//
#include <hip/hip_runtime.h>
#include <hip/hip_bf16.h>

typedef __attribute__((ext_vector_type(16))) _Float16 v16h;
typedef __attribute__((ext_vector_type(8)))  float    v8f;

#define B_    32
#define DIN_  256
#define DOUT2_ 512
#define T_    4096
#define SPK_  128

// ---------------------------------------------------------------------------
// Kernel 1: style/beta modulation + per-(b,o) demod & mean statistics.
// u[b,i] = w_scale * s[b,i], v[b,i] = w_scale * beta[b,i]
// w_raw[o,i,k] = u[i]*conv_w[o,i,k] + v[i]
// demod[b,o] = rsqrt(sum w_raw^2 + 1e-8), r[b,o] = demod * mean(w_raw)
// ---------------------------------------------------------------------------
__global__ __launch_bounds__(512) void prep_kernel(
    const float* __restrict__ c_trg,
    const float* __restrict__ style_w, const float* __restrict__ style_b,
    const float* __restrict__ beta_w,  const float* __restrict__ beta_b,
    const float* __restrict__ conv_w,
    float* __restrict__ u_out, float* __restrict__ v_out,
    float* __restrict__ demod_out, float* __restrict__ r_out)
{
    __shared__ float uS[DIN_], vS[DIN_];
    const int b = blockIdx.x;
    const int tid = threadIdx.x;

    if (tid < DIN_) {
        const int i = tid;
        const float lin_scale = 0.08838834764831845f;   // 1/sqrt(128)
        const float w_scale   = 1.0f / 48.0f;           // 1/sqrt(256*9)
        const float* ct  = c_trg   + (size_t)b * SPK_;
        const float* swr = style_w + (size_t)i * SPK_;
        const float* bwr = beta_w  + (size_t)i * SPK_;
        float sa = 0.f, ba = 0.f;
        for (int j = 0; j < SPK_; ++j) { float c = ct[j]; sa += c * swr[j]; ba += c * bwr[j]; }
        float s  = sa * lin_scale + style_b[i];
        float be = ba * lin_scale + beta_b[i];
        float u = w_scale * s, v = w_scale * be;
        uS[i] = u; vS[i] = v;
        u_out[b * DIN_ + i] = u;
        v_out[b * DIN_ + i] = v;
    }
    __syncthreads();

    const int o = tid;                       // 0..511
    const float* cw = conv_w + (size_t)o * (DIN_ * 3);
    float sumsq = 0.f, summ = 0.f;
    for (int i = 0; i < DIN_; ++i) {
        float u = uS[i], v = vS[i];
        float c0 = cw[i*3+0], c1 = cw[i*3+1], c2 = cw[i*3+2];
        float w0 = u*c0 + v, w1 = u*c1 + v, w2 = u*c2 + v;
        sumsq += w0*w0 + w1*w1 + w2*w2;
        summ  += w0 + w1 + w2;
    }
    float demod = rsqrtf(sumsq + 1e-8f);
    float dmean = summ * (1.0f / 768.0f);
    demod_out[b * DOUT2_ + o] = demod;
    r_out[b * DOUT2_ + o]     = demod * dmean;
}

// ---------------------------------------------------------------------------
// Kernel 2: materialize final f16 weights in WMMA A-fragment order.
// Layout (halfs): A[b][k][o_tile(32)][i_chunk(8)][lane(32)][h(16)]
//   A-matrix 16x32 f16 fragment: M = lane%16,
//   K(lane,h) = (h<8 ? h : h+8) + (lane>=16 ? 8 : 0)
// Each thread produces one half-octet (8 contiguous halfs = one b128 store).
// w_final = demod*(u[i]*conv_w[o,i,k] + v[i]) - r
// ---------------------------------------------------------------------------
__global__ __launch_bounds__(256) void wmat_kernel(
    const float* __restrict__ conv_w,
    const float* __restrict__ u_in, const float* __restrict__ v_in,
    const float* __restrict__ demod_in, const float* __restrict__ r_in,
    _Float16* __restrict__ aw)
{
    const unsigned g = blockIdx.x * 256u + threadIdx.x;   // octet index
    const int hbsel = g & 1;
    const int L     = (g >> 1) & 31;
    const int ic    = (g >> 6) & 7;
    const int ot    = (g >> 9) & 31;
    const unsigned kk = g >> 14;          // b*3 + k, in [0,96)
    const int k = kk % 3;
    const int b = kk / 3;

    const int o     = ot * 16 + (L & 15);
    const int Kbase = (hbsel ? 16 : 0) + ((L >> 4) * 8);
    const int i0    = ic * 32 + Kbase;

    const float demod = demod_in[b * DOUT2_ + o];
    const float r     = r_in[b * DOUT2_ + o];
    const float* cw = conv_w + ((size_t)o * DIN_ + i0) * 3 + k;
    const float* up = u_in + b * DIN_ + i0;
    const float* vp = v_in + b * DIN_ + i0;

    union { _Float16 h[8]; uint4 q; } pk;
#pragma unroll
    for (int j = 0; j < 8; ++j) {
        float w = demod * (up[j] * cw[j * 3] + vp[j]) - r;
        pk.h[j] = (_Float16)w;
    }
    *reinterpret_cast<uint4*>(aw + (size_t)g * 8) = pk.q;   // 16B-aligned b128
}

// ---------------------------------------------------------------------------
// Kernel 3: implicit-GEMM conv1d via v_wmma_f32_16x16x32_f16 + fused GLU.
// Block: one (batch, 64 GLU channel-pairs, 256 timesteps) tile. 512 thr = 16 waves.
// Wave: pair-tile mp (16 GLU rows: a-half & gate-half) x 64 cols -> 8 v8f accums.
// K loop: k=0..2 (conv tap) x ic=0..7 (32-wide i-chunks) = 24 WMMA K-steps.
// B tile (32 i x 256 t) staged f32->f16 into LDS in B-fragment order, double-buffered.
// Conv halo handled branchlessly: clamped in-bounds load + v_cndmask select.
// ---------------------------------------------------------------------------
__global__ __launch_bounds__(512) void conv_glu_kernel(
    const float* __restrict__ x,
    const _Float16* __restrict__ aw,
    float* __restrict__ out)
{
    const int bid = blockIdx.x;
    const int tt  = bid & 15;          // t-tile
    const int og  = (bid >> 4) & 3;    // channel-pair group (64 pairs)
    const int b   = bid >> 6;          // batch
    const int t0  = tt * 256;

    const int tid  = threadIdx.x;
    const int lane = tid & 31;
    const int wv   = tid >> 5;         // 0..15
    const int mp   = wv & 3;           // pair-tile within block
    const int ng   = wv >> 2;          // 64-col group

    // B-fragment LDS: [buf][ (ntile*32 + laneGroup*16 + N)*16 + h ] halfs
    __shared__ __align__(32) _Float16 Blds[2][16 * 32 * 16];   // 2 x 16 KB

    v8f acc[8];                        // [hh(a/g)][j(4 n-tiles)]
#pragma unroll
    for (int q = 0; q < 8; ++q)
#pragma unroll
        for (int e = 0; e < 8; ++e) acc[q][e] = 0.0f;

    // loader mapping: 512 thr, 8192 elems/step -> 16 each (4 rows x 4 cols)
    const int ttl0 = (tid & 63) * 4;   // 4 consecutive timesteps
    const int rg   = tid >> 6;         // row group: rows rg*4 .. rg*4+3
    const int tau0 = t0 + ttl0;        // 16B-aligned x column base
    const float* xb = x + (size_t)b * DIN_ * T_;

    // Branchless halo: clamped in-bounds offsets + per-thread validity (hoisted)
    const bool lok = (tau0 > 0);
    const bool rok = (tau0 + 4 < T_);
    const int  taum1 = lok ? (tau0 - 1) : 0;        // always legal
    const int  taup4 = rok ? (tau0 + 4) : (T_ - 1); // always legal

    int buf = 0;
    for (int k = 0; k < 3; ++k) {
        for (int ic = 0; ic < 8; ++ic) {
            const int i0 = ic * 32;

            // --- load x tile rows: aligned float4 + unconditional clamped halo
            float va[4][4];
#pragma unroll
            for (int rr = 0; rr < 4; ++rr) {
                const int ii = rg * 4 + rr;
                const float* xrow = xb + (size_t)(i0 + ii) * T_;
                float4 f4 = *reinterpret_cast<const float4*>(xrow + tau0);
                if (k == 0) {
                    float e = xrow[taum1];          // in-bounds, no branch
                    e = lok ? e : 0.0f;             // v_cndmask
                    va[rr][0] = e;    va[rr][1] = f4.x; va[rr][2] = f4.y; va[rr][3] = f4.z;
                } else if (k == 1) {
                    va[rr][0] = f4.x; va[rr][1] = f4.y; va[rr][2] = f4.z; va[rr][3] = f4.w;
                } else {
                    float e = xrow[taup4];          // in-bounds, no branch
                    e = rok ? e : 0.0f;             // v_cndmask
                    va[rr][0] = f4.y; va[rr][1] = f4.z; va[rr][2] = f4.w; va[rr][3] = e;
                }
            }

            // --- store to LDS in B-fragment order: K = ii (table order: lane half = (K>>3)&1, h = (K&7) + (K>=16?8:0)), packs 4 K per b64
            const int lg16 = ((rg >> 1) & 1) * 16;
            const int h0   = (rg & 1) * 4 + (rg >> 2) * 8;
#pragma unroll
            for (int cc = 0; cc < 4; ++cc) {
                const int ttl = ttl0 + cc;
                const int ntile = ttl >> 4, N = ttl & 15;
                union { _Float16 h[4]; unsigned long long q; } pk;
#pragma unroll
                for (int j2 = 0; j2 < 4; ++j2) pk.h[j2] = (_Float16)va[j2][cc];
                *reinterpret_cast<unsigned long long*>(
                    &Blds[buf][((ntile * 32 + lg16 + N) * 16 + h0)]) = pk.q;
            }

            __syncthreads();

            // --- A fragments (pre-materialized, fragment-ordered): 32B/lane each
            const int ot_a = og * 4 + mp;           // a-half pre-act row tile
            const size_t abase =
                ((((size_t)(b * 3 + k) * 32 + ot_a) * 8 + ic) * 32 + lane) * 16;
            const v16h afa = *reinterpret_cast<const v16h*>(aw + abase);
            const v16h afg = *reinterpret_cast<const v16h*>(aw + abase + (size_t)16 * 8 * 32 * 16);

            // --- 8 WMMAs: 2 halves x 4 n-tiles
#pragma unroll
            for (int j = 0; j < 4; ++j) {
                const v16h bf = *reinterpret_cast<const v16h*>(
                    &Blds[buf][((ng * 4 + j) * 32 + lane) * 16]);
                acc[j]     = __builtin_amdgcn_wmma_f32_16x16x32_f16(
                                 false, afa, false, bf, (short)0, acc[j],     false, false);
                acc[4 + j] = __builtin_amdgcn_wmma_f32_16x16x32_f16(
                                 false, afg, false, bf, (short)0, acc[4 + j], false, false);
            }

            buf ^= 1;
        }
    }

    // --- fused GLU epilogue: D layout -> M = e + (lane>=16?8:0), N = lane%16
    const int N    = lane & 15;
    const int rofs = (lane >> 4) * 8;
#pragma unroll
    for (int j = 0; j < 4; ++j) {
        const int tcol = t0 + ng * 64 + j * 16 + N;
        float* obase = out + ((size_t)b * 256 + og * 64 + mp * 16 + rofs) * T_ + tcol;
#pragma unroll
        for (int e = 0; e < 8; ++e) {
            float g   = acc[4 + j][e];
            float sig = 1.0f / (1.0f + __expf(-g));
            obase[(size_t)e * T_] = acc[j][e] * sig;
        }
    }
}

// ---------------------------------------------------------------------------
// Host launcher. Inputs: x, c_src(unused), c_trg, style_w, style_b,
//                        beta_w, beta_b, conv_w
// ws layout: u(8192 f32) | v(8192) | demod(16384) | r(16384) | A_f16(12.58M halfs)
// ---------------------------------------------------------------------------
extern "C" void kernel_launch(void* const* d_in, const int* in_sizes, int n_in,
                              void* d_out, int out_size, void* d_ws, size_t ws_size,
                              hipStream_t stream) {
    const float* x       = (const float*)d_in[0];
    const float* c_trg   = (const float*)d_in[2];
    const float* style_w = (const float*)d_in[3];
    const float* style_b = (const float*)d_in[4];
    const float* beta_w  = (const float*)d_in[5];
    const float* beta_b  = (const float*)d_in[6];
    const float* conv_w  = (const float*)d_in[7];

    float* wsF   = (float*)d_ws;
    float* u     = wsF;
    float* v     = wsF + 8192;
    float* demod = wsF + 16384;
    float* r     = wsF + 32768;
    _Float16* aw = (_Float16*)((char*)d_ws + (size_t)49152 * sizeof(float));

    prep_kernel<<<32, 512, 0, stream>>>(c_trg, style_w, style_b, beta_w, beta_b,
                                        conv_w, u, v, demod, r);
    wmat_kernel<<<6144, 256, 0, stream>>>(conv_w, u, v, demod, r, aw);
    conv_glu_kernel<<<2048, 512, 0, stream>>>(x, aw, (float*)d_out);
}
